// MORRSdpa_33389075759468
// MI455X (gfx1250) — matrix operand order repaired and checked by reading, not compiled, not run on hardware
//
#include <hip/hip_runtime.h>

// ---------------------------------------------------------------------------
// Flash-attention SDPA for MI455X (gfx1250, wave32, WMMA).
// B=1, H=16, S=4096, D=64, fp32 in/out; bf16 WMMA with fp32 accumulation.
// Transposed formulation: S^T = K Q^T and O^T = V^T P^T, so softmax runs as
// per-lane register trees (q across lanes, keys across VGPRs) with a single
// ds_swizzle half-swap per reduction, and P^T never touches LDS.
// 256 threads = 8 waves/block, QTILE=128, KTILE=64: 16 v_wmma per wave-iter.
// ---------------------------------------------------------------------------

typedef __attribute__((ext_vector_type(16))) __bf16 v16bf;
typedef __attribute__((ext_vector_type(4)))  __bf16 v4bf;
typedef __attribute__((ext_vector_type(2)))  __bf16 v2bf;
typedef __attribute__((ext_vector_type(8)))  float  v8f;
typedef __attribute__((ext_vector_type(4)))  float  v4f;

#define S_LEN  4096
#define D_DIM  64
#define HEADS  16
#define QTILE  128         // Q rows per block (16 per wave, 8 waves)
#define KTILE  64          // keys per inner iteration
#define QKSTR  72          // padded LDS row stride (elements) for Q/K tiles
#define VSTR   72          // padded LDS row stride (elements) for V^T tile

union FragBF {
    v16bf    v;
    uint4    q[2];         // two 16-byte LDS loads fill one A/B fragment
    unsigned u[8];         // for register-built B operands
};
union F8 {
    v8f   v;
    v4f   h[2];
    float f[8];
};

// swap the two 16-lane halves of the wave (LDS pipe, no memory traffic)
__device__ __forceinline__ float swz16f(float v) {
    return __int_as_float(__builtin_amdgcn_ds_swizzle(__float_as_int(v), 0x401F));
}
__device__ __forceinline__ unsigned swz16u(unsigned v) {
    return (unsigned)__builtin_amdgcn_ds_swizzle((int)v, 0x401F);
}
// pack two f32 -> packed bf16 pair
__device__ __forceinline__ unsigned pkbf(float a, float b) {
    union { v2bf b2; unsigned u; } w;
    w.b2[0] = (__bf16)a;
    w.b2[1] = (__bf16)b;
    return w.u;
}

__global__ __launch_bounds__(256) void
MORRSdpa_flash_kernel(const float* __restrict__ Q,
                      const float* __restrict__ K,
                      const float* __restrict__ V,
                      const float* __restrict__ mask,
                      float* __restrict__ out) {
    __shared__ __bf16 sQ [QTILE * QKSTR];       // 18432 B
    __shared__ __bf16 sK [KTILE * QKSTR];       //  9216 B
    __shared__ __bf16 sVt[D_DIM * VSTR];        //  9216 B (V transposed)

    const int h    = blockIdx.x;            // h fastest -> mask/KV reuse in L2
    const int qb   = blockIdx.y * QTILE;
    const int tid  = threadIdx.x;
    const int wave = tid >> 5;
    const int lane = tid & 31;
    const int ln   = lane & 15;             // q column this lane owns
    const int hi   = lane >> 4;             // which 16-lane half

    const float* Qh = Q + (size_t)h * S_LEN * D_DIM;
    const float* Kh = K + (size_t)h * S_LEN * D_DIM;
    const float* Vh = V + (size_t)h * S_LEN * D_DIM;

    // ---- stage Q tile (128x64), fold 1/sqrt(D) into the bf16 conversion ---
    #pragma unroll
    for (int i = 0; i < 8; ++i) {
        int idx = tid + i * 256;            // 2048 float4 total
        int row = idx >> 4;
        int c4  = idx & 15;
        v4f f = *(const v4f*)(Qh + (size_t)(qb + row) * D_DIM + c4 * 4);
        f *= 0.125f;
        *(v4bf*)&sQ[row * QKSTR + c4 * 4] = __builtin_convertvector(f, v4bf);
    }
    __syncthreads();

    // ---- Q as B-operand (d x q): lane = q, K-dim = d ----------------------
    // B layout: lane group g holds K = chunk*32 + g*16 .. +15 of column q.
    FragBF qbf[2];
    #pragma unroll
    for (int chunk = 0; chunk < 2; ++chunk) {
        const __bf16* base = &sQ[(wave * 16 + ln) * QKSTR + chunk * 32 + hi * 16];
        qbf[chunk].q[0] = *(const uint4*)(base);
        qbf[chunk].q[1] = *(const uint4*)(base + 8);
    }

    v8f acc[4] = {};            // O^T accumulator: d = f*16 + r + 8*hi, q = ln
    float m_run = -1e30f;       // per-lane running max (this lane's q)
    float l_run = 0.0f;         // per-lane running sum

    const int   qrow = qb + wave * 16 + ln;
    const float* mrow = mask + (size_t)qrow * S_LEN;   // this q's mask row

    for (int kb = 0; kb < S_LEN; kb += KTILE) {
        __syncthreads();                     // protect K/V LDS reuse

        // ---- stage K tile row-major, V tile transposed ------------------
        #pragma unroll
        for (int i = 0; i < 4; ++i) {
            int idx = tid + i * 256;         // 1024 float4 per matrix
            int row = idx >> 4;
            int c4  = idx & 15;
            v4f fk = *(const v4f*)(Kh + (size_t)(kb + row) * D_DIM + c4 * 4);
            *(v4bf*)&sK[row * QKSTR + c4 * 4] = __builtin_convertvector(fk, v4bf);
            v4f fv = *(const v4f*)(Vh + (size_t)(kb + row) * D_DIM + c4 * 4);
            v4bf bv = __builtin_convertvector(fv, v4bf);
            sVt[(c4 * 4 + 0) * VSTR + row] = bv[0];
            sVt[(c4 * 4 + 1) * VSTR + row] = bv[1];
            sVt[(c4 * 4 + 2) * VSTR + row] = bv[2];
            sVt[(c4 * 4 + 3) * VSTR + row] = bv[3];
        }
        if (kb + KTILE < S_LEN) {            // warm caches for next tile
            const float* nk = Kh + (size_t)(kb + KTILE + (tid >> 2)) * D_DIM + (tid & 3) * 16;
            const float* nv = Vh + (size_t)(kb + KTILE + (tid >> 2)) * D_DIM + (tid & 3) * 16;
            __builtin_prefetch(nk, 0, 3);
            __builtin_prefetch(nv, 0, 3);
        }
        __syncthreads();

        // ---- mask: 8 consecutive floats per 16-key tile (vector loads) ---
        F8 mk[4];
        #pragma unroll
        for (int nb = 0; nb < 4; ++nb) {
            const float* mp = mrow + kb + nb * 16 + hi * 8;
            mk[nb].h[0] = *(const v4f*)(mp);
            mk[nb].h[1] = *(const v4f*)(mp + 4);
        }

        // ---- S^T tiles: A = K rows (keys x d), B = Q (d x q) -------------
        v8f c[4] = {};
        #pragma unroll
        for (int chunk = 0; chunk < 2; ++chunk) {
            #pragma unroll
            for (int nb = 0; nb < 4; ++nb) { // 16-key M-tiles
                FragBF kf;   // A: lane row = key nb*16+ln, K = d subrange
                const __bf16* base = &sK[(nb * 16 + ln) * QKSTR + chunk * 32 + hi * 16];
                kf.q[0] = *(const uint4*)(base);
                kf.q[1] = *(const uint4*)(base + 8);
                c[nb] = __builtin_amdgcn_wmma_f32_16x16x32_bf16(
                            false, kf.v, false, qbf[chunk].v, (short)0, c[nb], false, false);
            }
        }

        // ---- online softmax: this lane owns q=ln; keys live in registers -
        F8 x[4];
        #pragma unroll
        for (int nb = 0; nb < 4; ++nb) x[nb].v = c[nb] + mk[nb].v;

        float t = -1e30f;
        #pragma unroll
        for (int nb = 0; nb < 4; ++nb)
            #pragma unroll
            for (int r = 0; r < 8; ++r) t = fmaxf(t, x[nb].f[r]);
        t = fmaxf(t, swz16f(t));             // combine the two key-halves

        float mnew = fmaxf(m_run, t);
        float sc   = __expf(m_run - mnew);

        F8 p[4];
        float s = 0.0f;
        #pragma unroll
        for (int nb = 0; nb < 4; ++nb) {
            v8f e = x[nb].v - mnew;
            #pragma unroll
            for (int r = 0; r < 8; ++r) {
                float pe = __expf(e[r]);
                p[nb].f[r] = pe;
                s += pe;
            }
        }
        s += swz16f(s);                      // other half's partial sum
        l_run = l_run * sc + s;
        m_run = mnew;

        #pragma unroll
        for (int f = 0; f < 4; ++f) acc[f] *= sc;

        // ---- pack P^T (C layout) into bf16 pairs ------------------------
        // pk[nb][v] = keys nb*16 + 8*hi + {2v,2v+1} of column q (this lane)
        unsigned pk[4][4];
        #pragma unroll
        for (int nb = 0; nb < 4; ++nb)
            #pragma unroll
            for (int v = 0; v < 4; ++v)
                pk[nb][v] = pkbf(p[nb].f[2 * v], p[nb].f[2 * v + 1]);

        // ---- O^T += V^T P^T : A = V^T rows (d x keys), B = P^T ----------
        #pragma unroll
        for (int ck = 0; ck < 2; ++ck) {
            // Build B operand (keys x q): lane group g needs tile 2ck+g.
            // B VGPR w holds in-tile keys {2w,2w+1}: w<4 sourced from the
            // hi=0 half, w>=4 from the hi=1 half -> one half-swap each.
            FragBF pb;
            #pragma unroll
            for (int w = 0; w < 4; ++w) {
                unsigned other = swz16u(pk[2 * ck + 1][w]);
                pb.u[w] = hi ? other : pk[2 * ck][w];
            }
            #pragma unroll
            for (int w = 0; w < 4; ++w) {
                unsigned other = swz16u(pk[2 * ck][w]);
                pb.u[4 + w] = hi ? pk[2 * ck + 1][w] : other;
            }
            #pragma unroll
            for (int f = 0; f < 4; ++f) {    // A: lane row d = f*16+ln
                FragBF va;
                const __bf16* base = &sVt[(f * 16 + ln) * VSTR + ck * 32 + hi * 16];
                va.q[0] = *(const uint4*)(base);
                va.q[1] = *(const uint4*)(base + 8);
                acc[f] = __builtin_amdgcn_wmma_f32_16x16x32_bf16(
                             false, va.v, false, pb.v, (short)0, acc[f], false, false);
            }
        }
    }

    // ---- normalize and write out: lane writes its q row, vectorized ------
    float inv = 1.0f / l_run;
    float* opq = out + ((size_t)h * S_LEN + qrow) * D_DIM;
    #pragma unroll
    for (int f = 0; f < 4; ++f) {
        v4f o0, o1;
        #pragma unroll
        for (int j = 0; j < 4; ++j) { o0[j] = acc[f][j] * inv; o1[j] = acc[f][4 + j] * inv; }
        *(v4f*)(opq + f * 16 + hi * 8)     = o0;   // d = f*16 + 8*hi + 0..3
        *(v4f*)(opq + f * 16 + hi * 8 + 4) = o1;   // d = f*16 + 8*hi + 4..7
    }
}

extern "C" void kernel_launch(void* const* d_in, const int* in_sizes, int n_in,
                              void* d_out, int out_size, void* d_ws, size_t ws_size,
                              hipStream_t stream) {
    (void)in_sizes; (void)n_in; (void)out_size; (void)d_ws; (void)ws_size;
    const float* Q    = (const float*)d_in[0];
    const float* K    = (const float*)d_in[1];
    const float* V    = (const float*)d_in[2];
    const float* mask = (const float*)d_in[3];
    float* out = (float*)d_out;

    dim3 grid(HEADS, S_LEN / QTILE);   // h fastest-varying: L2 reuse of mask/KV
    MORRSdpa_flash_kernel<<<grid, 256, 0, stream>>>(Q, K, V, mask, out);
}
